// fliicbsimpleKT_28544352649373
// MI455X (gfx1250) — compile-verified, matrix operand order checked by reading, not run on hardware
//
#include <hip/hip_runtime.h>
#include <hip/hip_bf16.h>

// ---------------------------------------------------------------------------
// Transformer forward (B=16,S=512,D=1024,H=16,DK=64,F=4096,L=4) on gfx1250.
// Compute-bound (~824 GFLOP vs ~12us of HBM traffic at 23.3 TB/s) -> all GEMMs
// on v_wmma_f32_16x16x32_bf16. All WMMA fragments are loaded from LDS as
// contiguous ds_load_b128 (weights pre-transposed; V written transposed).
// ---------------------------------------------------------------------------

typedef __attribute__((ext_vector_type(16))) __bf16 v16bf;
typedef __attribute__((ext_vector_type(8)))  float  v8f;

#define CB  16
#define CS  512
#define CD  1024
#define CH  16
#define CDK 64
#define CF  4096
#define CL  4
#define NTOK (CB * CS)   // 8192

__device__ __forceinline__ v8f vzero8() {
  v8f z = {0.f,0.f,0.f,0.f,0.f,0.f,0.f,0.f};
  return z;
}

__device__ __forceinline__ v8f wmma_bf16(v16bf a, v16bf b, v8f c) {
  return __builtin_amdgcn_wmma_f32_16x16x32_bf16(false, a, false, b,
                                                 (short)0, c, false, false);
}

// 16x32 bf16 fragment from a row-major (16 x >=32) LDS tile; contiguous rows,
// so this lowers to 2x ds_load_b128 per lane. Serves both the A operand and
// (via transposed-tile staging) the B operand.
__device__ __forceinline__ v16bf load_frag(const __bf16* p, int stride, int lane) {
  int m = lane & 15, h = lane >> 4;
  const __bf16* r = p + m * stride + h * 8;
  v16bf f;
#pragma unroll
  for (int i = 0; i < 8; ++i) { f[i] = r[i]; f[8 + i] = r[16 + i]; }
  return f;
}

// ---------------------------------------------------------------------------
// Elementwise / transpose helpers
// ---------------------------------------------------------------------------
__global__ void k_add_pe(const float* __restrict__ q, const float* __restrict__ qa,
                         const float* __restrict__ pe,
                         float* __restrict__ xF, __bf16* __restrict__ xB,
                         __bf16* __restrict__ yB, int total, int SD) {
  int i = blockIdx.x * blockDim.x + threadIdx.x;
  if (i >= total) return;
  int sd = i % SD;
  float p  = pe[sd];
  float xv = q[i] + p;
  xF[i] = xv;
  xB[i] = (__bf16)xv;
  yB[i] = (__bf16)(qa[i] + p);
}

// Tiled transpose + f32->bf16: in [z][R][C] f32 -> out [z][C][R] bf16.
__global__ __launch_bounds__(256)
void k_transpose_f2bf(const float* __restrict__ in, __bf16* __restrict__ out,
                      int R, int C) {
  __shared__ float tile[32][33];
  const size_t zo = (size_t)blockIdx.z * R * C;
  in  += zo;
  out += zo;
  int bx = blockIdx.x * 32;           // col base (C)
  int by = blockIdx.y * 32;           // row base (R)
  int tx = threadIdx.x & 31, ty = threadIdx.x >> 5;   // 32 x 8
#pragma unroll
  for (int j = 0; j < 32; j += 8)
    tile[ty + j][tx] = in[(size_t)(by + ty + j) * C + bx + tx];
  __syncthreads();
#pragma unroll
  for (int j = 0; j < 32; j += 8)
    out[(size_t)(bx + ty + j) * R + by + tx] = (__bf16)tile[tx][ty + j];
}

// ---------------------------------------------------------------------------
// bf16 GEMM: out[M,N] = A[M,K] @ W[K,N] + bias (+ReLU), W given TRANSPOSED
// (WT is N x K). Block 128x128x32, 256 threads (8 waves), wave tile 64x32.
// Double-buffered LDS; every fragment load is 2x ds_load_b128.
// Optional outputs: f32 row-major, bf16 row-major, bf16 transposed (N x M).
// ---------------------------------------------------------------------------
#define GBM 128
#define GBN 128
#define GBK 32
#define TSTR (GBK + 8)    // 40

__global__ __launch_bounds__(256)
void k_gemm_bf16(const __bf16* __restrict__ A, const __bf16* __restrict__ WT,
                 const float* __restrict__ bias,
                 float* __restrict__ outF, __bf16* __restrict__ outB,
                 __bf16* __restrict__ outBT,
                 int M, int N, int K, int relu) {
  __shared__ __bf16 As[2][GBM * TSTR];   // 128 rows (m) x 32 k
  __shared__ __bf16 Bs[2][GBN * TSTR];   // 128 rows (n) x 32 k  (= W^T tile)

  const int tid  = threadIdx.x;
  const int lane = tid & 31;
  const int wave = tid >> 5;
  const int wm   = (wave >> 2) * 64;   // 0 / 64
  const int wn   = (wave & 3) * 32;    // 0 / 32 / 64 / 96
  const int bRow = blockIdx.y * GBM;
  const int bCol = blockIdx.x * GBN;

  v8f acc[4][2];
#pragma unroll
  for (int i = 0; i < 4; ++i)
#pragma unroll
    for (int j = 0; j < 2; ++j) acc[i][j] = vzero8();

  auto stage = [&](int k0, int p) {
#pragma unroll
    for (int c = tid; c < 512; c += 256) {   // 512 chunks of 8 bf16 per tile
      int r = c >> 2, cc = (c & 3) * 8;
      uint4 va = *reinterpret_cast<const uint4*>(A  + (size_t)(bRow + r) * K + k0 + cc);
      uint4 vb = *reinterpret_cast<const uint4*>(WT + (size_t)(bCol + r) * K + k0 + cc);
      const __bf16* pa = reinterpret_cast<const __bf16*>(&va);
      const __bf16* pb = reinterpret_cast<const __bf16*>(&vb);
#pragma unroll
      for (int j = 0; j < 8; ++j) {
        As[p][r * TSTR + cc + j] = pa[j];
        Bs[p][r * TSTR + cc + j] = pb[j];
      }
    }
  };

  stage(0, 0);
  int p = 0;
  for (int k0 = 0; k0 < K; k0 += GBK) {
    __syncthreads();
    if (k0 + GBK < K) stage(k0 + GBK, p ^ 1);
    if (k0 + 2 * GBK < K) {
      __builtin_prefetch(A  + (size_t)(bRow + (tid >> 1)) * K + k0 + 2 * GBK, 0, 0);
      __builtin_prefetch(WT + (size_t)(bCol + (tid >> 1)) * K + k0 + 2 * GBK, 0, 0);
    }
    v16bf af[4];
#pragma unroll
    for (int ms = 0; ms < 4; ++ms)
      af[ms] = load_frag(&As[p][(wm + ms * 16) * TSTR], TSTR, lane);
    v16bf bfr[2];
#pragma unroll
    for (int ns = 0; ns < 2; ++ns)
      bfr[ns] = load_frag(&Bs[p][(wn + ns * 16) * TSTR], TSTR, lane);
#pragma unroll
    for (int ms = 0; ms < 4; ++ms)
#pragma unroll
      for (int ns = 0; ns < 2; ++ns)
        acc[ms][ns] = wmma_bf16(af[ms], bfr[ns], acc[ms][ns]);
    p ^= 1;
  }

  // C layout: lane = (m/8)*16 + n, vgpr i = m%8 -> each lane owns 8
  // consecutive rows of one column (perfect for the transposed bf16 store).
  const int n  = lane & 15;
  const int mb = (lane >> 4) * 8;
#pragma unroll
  for (int ms = 0; ms < 4; ++ms)
#pragma unroll
    for (int ns = 0; ns < 2; ++ns) {
      int row = bRow + wm + ms * 16 + mb;
      int col = bCol + wn + ns * 16 + n;
      float bs = bias[col];
      alignas(16) __bf16 tb[8];
#pragma unroll
      for (int i = 0; i < 8; ++i) {
        float v = acc[ms][ns][i] + bs;
        if (relu) v = fmaxf(v, 0.f);
        tb[i] = (__bf16)v;
        size_t idx = (size_t)(row + i) * N + col;
        if (outF) outF[idx] = v;
        if (outB) outB[idx] = (__bf16)v;
      }
      if (outBT)   // one 16B store: rows row..row+7 at column col
        *reinterpret_cast<uint4*>(outBT + (size_t)col * M + row) =
            *reinterpret_cast<const uint4*>(tb);
    }
}

// ---------------------------------------------------------------------------
// Flash attention, causal strictly-lower mask, query row 0 zeroed.
// Grid (S/64, H, B), 128 threads (4 waves); wave owns a 16-row q stripe.
// V arrives TRANSPOSED ([D][NTOK]) so the P@V B operand reads contiguously.
// ---------------------------------------------------------------------------
#define QSTR 72

__global__ __launch_bounds__(128)
void k_attn(const __bf16* __restrict__ QK, const __bf16* __restrict__ VT,
            __bf16* __restrict__ O) {
  __shared__ __bf16 Qs[64 * QSTR];
  __shared__ __bf16 Ks[64 * QSTR];   // [kv][d]   (row-major K == K^T operand)
  __shared__ __bf16 Vs[64 * QSTR];   // [d][kv]   (transposed V tile)
  __shared__ __bf16 Ps[64 * QSTR];   // [q][kv]

  const int qt = blockIdx.x, h = blockIdx.y, b = blockIdx.z;
  const int tid = threadIdx.x, lane = tid & 31, wave = tid >> 5;
  const int hc = h * CDK;
  const int q0 = qt * 64;
  const int wr = wave * 16;
  const int n  = lane & 15;
  const int mb = (lane >> 4) * 8;

  for (int e = tid; e < 512; e += 128) {
    int r = e >> 3, c = (e & 7) * 8;
    uint4 v = *reinterpret_cast<const uint4*>(
        QK + ((size_t)(b * CS + q0 + r)) * CD + hc + c);
    const __bf16* pv = reinterpret_cast<const __bf16*>(&v);
#pragma unroll
    for (int j = 0; j < 8; ++j) Qs[r * QSTR + c + j] = pv[j];
  }
  __syncthreads();

  v16bf aq0 = load_frag(&Qs[wr * QSTR], QSTR, lane);
  v16bf aq1 = load_frag(&Qs[wr * QSTR + 32], QSTR, lane);

  v8f cacc[4];
#pragma unroll
  for (int i = 0; i < 4; ++i) cacc[i] = vzero8();
  float rm[8], rl[8];
#pragma unroll
  for (int i = 0; i < 8; ++i) { rm[i] = -3.0e38f; rl[i] = 0.f; }

  const float scale = 0.125f;  // 1/sqrt(64)

  for (int t = 0; t <= qt; ++t) {
    __syncthreads();
    for (int e = tid; e < 512; e += 128) {
      int r = e >> 3, c = (e & 7) * 8;
      uint4 kv = *reinterpret_cast<const uint4*>(
          QK + ((size_t)(b * CS + t * 64 + r)) * CD + hc + c);          // [kv][d]
      uint4 vv = *reinterpret_cast<const uint4*>(
          VT + ((size_t)(hc + r)) * NTOK + (size_t)b * CS + t * 64 + c); // [d][kv]
      const __bf16* pk = reinterpret_cast<const __bf16*>(&kv);
      const __bf16* pv = reinterpret_cast<const __bf16*>(&vv);
#pragma unroll
      for (int j = 0; j < 8; ++j) {
        Ks[r * QSTR + c + j] = pk[j];
        Vs[r * QSTR + c + j] = pv[j];
      }
    }
    __syncthreads();

    // scores: Q(16x64) @ K^T -- row-major K tile IS the transposed operand
    v8f s[4];
#pragma unroll
    for (int nt = 0; nt < 4; ++nt) {
      v8f a = vzero8();
      v16bf b0 = load_frag(&Ks[(nt * 16) * QSTR], QSTR, lane);
      v16bf b1 = load_frag(&Ks[(nt * 16) * QSTR + 32], QSTR, lane);
      a = wmma_bf16(aq0, b0, a);
      a = wmma_bf16(aq1, b1, a);
      s[nt] = a;
    }
#pragma unroll
    for (int nt = 0; nt < 4; ++nt)
#pragma unroll
      for (int i = 0; i < 8; ++i) {
        float v = s[nt][i] * scale;
        if (t == qt && !((nt * 16 + n) < (wr + mb + i))) v = -1e32f;
        s[nt][i] = v;
      }
    // online softmax; rows replicate across the 16-lane groups
#pragma unroll
    for (int i = 0; i < 8; ++i) {
      float v = fmaxf(fmaxf(s[0][i], s[1][i]), fmaxf(s[2][i], s[3][i]));
#pragma unroll
      for (int off = 8; off >= 1; off >>= 1) v = fmaxf(v, __shfl_xor(v, off, 32));
      float mnew = fmaxf(rm[i], v);
      float corr = __expf(rm[i] - mnew);
      float psum = 0.f;
#pragma unroll
      for (int nt = 0; nt < 4; ++nt) {
        float pp = __expf(s[nt][i] - mnew);
        s[nt][i] = pp;
        psum += pp;
      }
#pragma unroll
      for (int off = 8; off >= 1; off >>= 1) psum += __shfl_xor(psum, off, 32);
      rl[i] = rl[i] * corr + psum;
      rm[i] = mnew;
#pragma unroll
      for (int nt = 0; nt < 4; ++nt) cacc[nt][i] *= corr;
    }
    // P (C layout) -> LDS row-major [q][kv]; wave-private rows, in-order LDS
#pragma unroll
    for (int nt = 0; nt < 4; ++nt)
#pragma unroll
      for (int i = 0; i < 8; ++i)
        Ps[(wr + mb + i) * QSTR + nt * 16 + n] = (__bf16)s[nt][i];

    v16bf ap0 = load_frag(&Ps[wr * QSTR], QSTR, lane);
    v16bf ap1 = load_frag(&Ps[wr * QSTR + 32], QSTR, lane);
#pragma unroll
    for (int nt = 0; nt < 4; ++nt) {
      v16bf bv0 = load_frag(&Vs[(nt * 16) * QSTR], QSTR, lane);        // kv 0..31
      v16bf bv1 = load_frag(&Vs[(nt * 16) * QSTR + 32], QSTR, lane);   // kv 32..63
      cacc[nt] = wmma_bf16(ap0, bv0, cacc[nt]);
      cacc[nt] = wmma_bf16(ap1, bv1, cacc[nt]);
    }
  }

#pragma unroll
  for (int nt = 0; nt < 4; ++nt)
#pragma unroll
    for (int i = 0; i < 8; ++i) {
      int qg = q0 + wr + mb + i;
      float v = (qg == 0) ? 0.f : cacc[nt][i] / rl[i];
      O[((size_t)(b * CS) + qg) * CD + hc + nt * 16 + n] = (__bf16)v;
    }
}

// ---------------------------------------------------------------------------
// Residual + LayerNorm over D=1024; one block (256 thr) per token row.
// ---------------------------------------------------------------------------
__global__ __launch_bounds__(256)
void k_res_ln(const float* __restrict__ X, const float* __restrict__ R,
              const float* __restrict__ g, const float* __restrict__ bta,
              float* __restrict__ outF, __bf16* __restrict__ outB) {
  __shared__ float red[256];
  const int row = blockIdx.x;
  const int tid = threadIdx.x;
  const float* xr = X + (size_t)row * CD;
  const float* rr = R + (size_t)row * CD;

  float vals[4];
  float s = 0.f;
#pragma unroll
  for (int j = 0; j < 4; ++j) {
    vals[j] = xr[tid + j * 256] + rr[tid + j * 256];
    s += vals[j];
  }
  red[tid] = s;
  __syncthreads();
  for (int st = 128; st > 0; st >>= 1) {
    if (tid < st) red[tid] += red[tid + st];
    __syncthreads();
  }
  float mu = red[0] / (float)CD;
  __syncthreads();

  float s2 = 0.f;
#pragma unroll
  for (int j = 0; j < 4; ++j) {
    float d = vals[j] - mu;
    s2 += d * d;
  }
  red[tid] = s2;
  __syncthreads();
  for (int st = 128; st > 0; st >>= 1) {
    if (tid < st) red[tid] += red[tid + st];
    __syncthreads();
  }
  float inv = rsqrtf(red[0] / (float)CD + 1e-5f);

#pragma unroll
  for (int j = 0; j < 4; ++j) {
    int c = tid + j * 256;
    float o = (vals[j] - mu) * inv * g[c] + bta[c];
    outF[(size_t)row * CD + c] = o;
    outB[(size_t)row * CD + c] = (__bf16)o;
  }
}

// ---------------------------------------------------------------------------
// Host orchestration
// ---------------------------------------------------------------------------
extern "C" void kernel_launch(void* const* d_in, const int* in_sizes, int n_in,
                              void* d_out, int out_size, void* d_ws, size_t ws_size,
                              hipStream_t stream) {
  (void)in_sizes; (void)n_in; (void)out_size; (void)ws_size;

  const float* q_embed = (const float*)d_in[0];
  const float* qa_embed = (const float*)d_in[1];
  const float* pe = (const float*)d_in[2];
  const float* Wk = (const float*)d_in[3];
  const float* bk = (const float*)d_in[4];
  const float* Wv = (const float*)d_in[5];
  const float* bv = (const float*)d_in[6];
  const float* Wo = (const float*)d_in[7];
  const float* bo = (const float*)d_in[8];
  const float* ln1_g = (const float*)d_in[9];
  const float* ln1_b = (const float*)d_in[10];
  const float* W1 = (const float*)d_in[11];
  const float* b1 = (const float*)d_in[12];
  const float* W2 = (const float*)d_in[13];
  const float* b2 = (const float*)d_in[14];
  const float* ln2_g = (const float*)d_in[15];
  const float* ln2_b = (const float*)d_in[16];

  char* ws = (char*)d_ws;
  size_t off = 0;
  auto alloc = [&](size_t bytes) -> void* {
    void* p = ws + off;
    off = (off + bytes + 255) & ~(size_t)255;
    return p;
  };

  float*  x_f32   = (float*) alloc((size_t)NTOK * CD * 4);
  __bf16* x_bf    = (__bf16*)alloc((size_t)NTOK * CD * 2);
  __bf16* y_bf    = (__bf16*)alloc((size_t)NTOK * CD * 2);
  __bf16* qk_bf   = (__bf16*)alloc((size_t)NTOK * CD * 2);
  __bf16* vT_bf   = (__bf16*)alloc((size_t)CD * NTOK * 2);   // transposed [D][NTOK]
  __bf16* ctx_bf  = (__bf16*)alloc((size_t)NTOK * CD * 2);
  float*  proj_f  = (float*) alloc((size_t)NTOK * CD * 4);
  __bf16* h_bf    = (__bf16*)alloc((size_t)NTOK * CF * 2);
  float*  ffn_f   = (float*) alloc((size_t)NTOK * CD * 4);
  __bf16* WkT     = (__bf16*)alloc((size_t)CL * CD * CD * 2);
  __bf16* WvT     = (__bf16*)alloc((size_t)CL * CD * CD * 2);
  __bf16* WoT     = (__bf16*)alloc((size_t)CL * CD * CD * 2);
  __bf16* W1T     = (__bf16*)alloc((size_t)CL * CD * CF * 2);
  __bf16* W2T     = (__bf16*)alloc((size_t)CL * CF * CD * 2);

  // weights -> bf16, transposed to [N][K] (deterministic each call)
  {
    dim3 gDD(CD / 32, CD / 32, CL);
    k_transpose_f2bf<<<gDD, 256, 0, stream>>>(Wk, WkT, CD, CD);
    k_transpose_f2bf<<<gDD, 256, 0, stream>>>(Wv, WvT, CD, CD);
    k_transpose_f2bf<<<gDD, 256, 0, stream>>>(Wo, WoT, CD, CD);
    k_transpose_f2bf<<<dim3(CF / 32, CD / 32, CL), 256, 0, stream>>>(W1, W1T, CD, CF);
    k_transpose_f2bf<<<dim3(CD / 32, CF / 32, CL), 256, 0, stream>>>(W2, W2T, CF, CD);
  }

  // x = q + pe ; y = qa + pe
  {
    int total = NTOK * CD;
    k_add_pe<<<dim3(total / 256), 256, 0, stream>>>(q_embed, qa_embed, pe,
                                                    x_f32, x_bf, y_bf,
                                                    total, CS * CD);
  }

  const dim3 gD(CD / GBN, NTOK / GBM);   // (8, 64)
  const dim3 gF(CF / GBN, NTOK / GBM);   // (32, 64)
  const dim3 gAttn(CS / 64, CH, CB);     // (8, 16, 16)

  for (int l = 0; l < CL; ++l) {
    const size_t oDD = (size_t)l * CD * CD;
    const size_t oDF = (size_t)l * CD * CF;

    // q/k projection (kq_same); v projection written TRANSPOSED
    k_gemm_bf16<<<gD, 256, 0, stream>>>(x_bf, WkT + oDD, bk + l * CD,
                                        nullptr, qk_bf, nullptr,
                                        NTOK, CD, CD, 0);
    k_gemm_bf16<<<gD, 256, 0, stream>>>(y_bf, WvT + oDD, bv + l * CD,
                                        nullptr, nullptr, vT_bf,
                                        NTOK, CD, CD, 0);
    // fused causal attention
    k_attn<<<gAttn, 128, 0, stream>>>(qk_bf, vT_bf, ctx_bf);
    // output projection
    k_gemm_bf16<<<gD, 256, 0, stream>>>(ctx_bf, WoT + oDD, bo + l * CD,
                                        proj_f, nullptr, nullptr,
                                        NTOK, CD, CD, 0);
    // x = LN(x + proj)
    k_res_ln<<<dim3(NTOK), 256, 0, stream>>>(x_f32, proj_f,
                                             ln1_g + l * CD, ln1_b + l * CD,
                                             x_f32, x_bf);
    // FFN
    k_gemm_bf16<<<gF, 256, 0, stream>>>(x_bf, W1T + oDF, b1 + l * CF,
                                        nullptr, h_bf, nullptr,
                                        NTOK, CF, CD, 1);
    k_gemm_bf16<<<gD, 256, 0, stream>>>(h_bf, W2T + oDF, b2 + l * CD,
                                        ffn_f, nullptr, nullptr,
                                        NTOK, CD, CF, 0);
    // x = LN(x + ffn)  (last layer writes straight to d_out)
    float* outF = (l == CL - 1) ? (float*)d_out : x_f32;
    k_res_ln<<<dim3(NTOK), 256, 0, stream>>>(x_f32, ffn_f,
                                             ln2_g + l * CD, ln2_b + l * CD,
                                             outF, x_bf);
  }
}
